// NaiveIndexer_72413148611123
// MI455X (gfx1250) — compile-verified, matrix operand order checked by reading, not run on hardware
//
#include <hip/hip_runtime.h>
#include <hip/hip_bf16.h>

#define TT    2048
#define HIDN  4096
#define RR    1536
#define NH    32
#define DD    128
#define TOPKK 1024
#define KLD   136   // padded LDS row stride (halfwords): 272B = 68 banks -> staggered

typedef __attribute__((ext_vector_type(16))) __bf16       v16bf;
typedef __attribute__((ext_vector_type(8)))  float        v8f;
typedef __attribute__((ext_vector_type(8)))  unsigned int v8u;
typedef __attribute__((ext_vector_type(4)))  int          v4i;

#define AS1 __attribute__((address_space(1)))
#define AS3 __attribute__((address_space(3)))

union FragCvt { v8u u; v16bf b; };

__device__ __forceinline__ unsigned short f2bf(float f) {
  unsigned int u = __float_as_uint(f);
  u += 0x7FFFu + ((u >> 16) & 1u);   // round-to-nearest-even
  return (unsigned short)(u >> 16);
}

// CDNA5 async copy: global -> LDS, 16B per lane, tracked by ASYNCcnt.
__device__ __forceinline__ void async_copy_b128(const void* g, void* l) {
#if __has_builtin(__builtin_amdgcn_global_load_async_to_lds_b128)
  __builtin_amdgcn_global_load_async_to_lds_b128((AS1 v4i*)g, (AS3 v4i*)l, 0, 0);
#else
  unsigned lds = (unsigned)(unsigned long long)l;
  asm volatile("global_load_async_to_lds_b128 %0, %1, off"
               :: "v"(lds), "v"(g) : "memory");
#endif
}

__device__ __forceinline__ void wait_async0() {
#if __has_builtin(__builtin_amdgcn_s_wait_asynccnt)
  __builtin_amdgcn_s_wait_asynccnt(0);
#else
  asm volatile("s_wait_asynccnt 0x0" ::: "memory");
#endif
}

// Load one 16x32 bf16 fragment (A-operand layout, ISA 7.12.2) from a
// row-major [rows x ld] bf16 matrix (global or LDS; compiler picks ds/vmem).
// Lane l (0..15): row = row0 + l; lanes 16..31: same rows, K shifted by 8.
// Half-elem e: K = e + (e>=8 ? 8 : 0) + (lane>=16 ? 8 : 0).
__device__ __forceinline__ v16bf load_frag_nt(const unsigned short* M,
                                              int ld, int row0, int k0, int lane) {
  const int row = row0 + (lane & 15);
  const int kb  = k0 + ((lane >> 4) << 3);
  const unsigned int* p = (const unsigned int*)(M + (size_t)row * ld + kb);
  FragCvt cv;
  cv.u[0] = p[0];  cv.u[1] = p[1];  cv.u[2] = p[2];   cv.u[3] = p[3];
  cv.u[4] = p[8];  cv.u[5] = p[9];  cv.u[6] = p[10];  cv.u[7] = p[11];
  return cv.b;
}

__global__ void __launch_bounds__(256)
cvt_f32_bf16(const float* __restrict__ in, unsigned short* __restrict__ out, int n) {
  int i = blockIdx.x * 256 + threadIdx.x;
  if (i < n) out[i] = f2bf(in[i]);
}

// C[M,N] = alpha * A[M,K] @ W[N,K]^T   (bf16 in, f32 out). One wave per 16x16 tile.
__global__ void __launch_bounds__(32)
gemm_nt_bf16_wmma(const unsigned short* __restrict__ A,
                  const unsigned short* __restrict__ W,
                  float* __restrict__ C,
                  int Ndim, int Kdim, float alpha) {
  const int lane = threadIdx.x;
  const int n0 = blockIdx.x * 16;
  const int m0 = blockIdx.y * 16;
  v8f acc = {};
  for (int k0 = 0; k0 < Kdim; k0 += 32) {
    if (k0 + 32 < Kdim) {  // pull next k-step into WGP-near caches (locality 3)
      __builtin_prefetch(A + (size_t)(m0 + (lane & 15)) * Kdim + k0 + 32, 0, 3);
      __builtin_prefetch(W + (size_t)(n0 + (lane & 15)) * Kdim + k0 + 32, 0, 3);
    }
    v16bf a = load_frag_nt(A, Kdim, m0, k0, lane);
    v16bf b = load_frag_nt(W, Kdim, n0, k0, lane);
    acc = __builtin_amdgcn_wmma_f32_16x16x32_bf16(false, a, false, b,
                                                  (short)0, acc, false, false);
  }
  // C/D layout: lane l, vgpr v -> row = v + (l>=16 ? 8 : 0), col = l & 15
  const int col = n0 + (lane & 15);
  const int r0  = m0 + ((lane >> 4) << 3);
#pragma unroll
  for (int v = 0; v < 8; ++v)
    C[(size_t)(r0 + v) * Ndim + col] = alpha * acc[v];
}

// RoPE on first 64 dims of each head, fold D^-0.5, emit bf16 q.
__global__ void __launch_bounds__(256)
q_post(const float* __restrict__ Qraw, const float* __restrict__ freqs,
       unsigned short* __restrict__ Qb) {
  const int t = blockIdx.x;
  for (int e = threadIdx.x; e < NH * DD; e += 256) {
    const int h = e >> 7, d = e & 127;
    const float* q = Qraw + (size_t)t * NH * DD + (size_t)h * DD;
    float o;
    if (d < 32) {
      float fr = freqs[t * 32 + d];
      o = q[d] * cosf(fr) - q[d + 32] * sinf(fr);
    } else if (d < 64) {
      int j = d - 32;
      float fr = freqs[t * 32 + j];
      o = q[j] * sinf(fr) + q[d] * cosf(fr);
    } else {
      o = q[d];
    }
    Qb[(size_t)t * NH * DD + e] = f2bf(o * 0.08838834764831845f); // * D^-0.5
  }
}

// layernorm over 128 -> RoPE first 64 -> bf16 k.
__global__ void __launch_bounds__(128)
k_post(const float* __restrict__ Kraw, const float* __restrict__ freqs,
       const float* __restrict__ gamma, const float* __restrict__ beta,
       unsigned short* __restrict__ Kb) {
  const int t = blockIdx.x;
  const int d = threadIdx.x;
  __shared__ float red[128];
  __shared__ float sh[128];
  float kv = Kraw[(size_t)t * DD + d];
  red[d] = kv; __syncthreads();
  for (int s = 64; s > 0; s >>= 1) { if (d < s) red[d] += red[d + s]; __syncthreads(); }
  float mu = red[0] * (1.0f / DD);
  __syncthreads();
  float c = kv - mu;
  red[d] = c * c; __syncthreads();
  for (int s = 64; s > 0; s >>= 1) { if (d < s) red[d] += red[d + s]; __syncthreads(); }
  float var = red[0] * (1.0f / DD);
  float nrm = c * rsqrtf(var + 1e-5f) * gamma[d] + beta[d];
  __syncthreads();
  sh[d] = nrm; __syncthreads();
  float o;
  if (d < 32)      { float fr = freqs[t*32+d];    o = sh[d] * cosf(fr) - sh[d+32] * sinf(fr); }
  else if (d < 64) { int j = d-32; float fr = freqs[t*32+j]; o = sh[j] * sinf(fr) + sh[d] * cosf(fr); }
  else             o = nrm;
  Kb[(size_t)t * DD + d] = f2bf(o);
}

// Fused: logits[t,s] = sum_h relu(q[t,h,:].k[s,:] * D^-0.5) * w[t,h] + mask.
// Block = 8 waves: one 16-row t-tile x 128-wide s-chunk. K-chunk staged to LDS
// once (async b128), per-head Q tile staged per iteration (async b128).
// LDS rows padded to KLD halfwords to stagger banks.
__global__ void __launch_bounds__(256)
logits_wmma(const unsigned short* __restrict__ Qb,   // [T, NH*DD]
            const unsigned short* __restrict__ Kb,   // [T, DD]
            const float* __restrict__ Wt,            // [T, NH]
            float* __restrict__ Lg) {                // [T, T]
  __shared__ unsigned short ksh[128 * KLD];  // ~34 KB
  __shared__ unsigned short qsh[16 * KLD];   // ~4.3 KB
  __shared__ float wsh[16 * NH];             // 2 KB
  const int tid  = threadIdx.x;
  const int lane = tid & 31;
  const int w    = tid >> 5;
  const int sC0  = blockIdx.x * 128;
  const int t0   = blockIdx.y * 16;

  // Stage K rows sC0..sC0+127 (contiguous in global; ld == DD) into padded LDS.
  {
    const unsigned short* gk = Kb + (size_t)sC0 * DD;
#pragma unroll
    for (int it = 0; it < 8; ++it) {
      int c = it * 256 + tid;                 // 16B chunk id: row = c/16, col-chunk = c%16
      int row = c >> 4, ch = c & 15;
      async_copy_b128(gk + (size_t)row * DD + ch * 8,
                      (void*)(ksh + row * KLD + ch * 8));
    }
  }
  for (int i = tid; i < 16 * NH; i += 256)
    wsh[i] = Wt[(size_t)(t0 + (i >> 5)) * NH + (i & 31)];
  wait_async0();
  __syncthreads();

  v16bf bk[4];                      // this wave's s-subtile, reused for all heads
#pragma unroll
  for (int kk = 0; kk < 4; ++kk)
    bk[kk] = load_frag_nt(ksh, KLD, w * 16, kk * 32, lane);

  float accL[8] = {};
  const int rloc = (lane >> 4) << 3;
  for (int h = 0; h < NH; ++h) {
    {  // stage Q[t0:t0+16, h*DD : h*DD+DD] -> qsh (16 B per thread)
      int row = tid >> 4, ch = tid & 15;
      const unsigned short* gq =
          Qb + (size_t)(t0 + row) * (NH * DD) + h * DD + ch * 8;
      async_copy_b128(gq, (void*)(qsh + row * KLD + ch * 8));
    }
    wait_async0();
    __syncthreads();
    v8f qk = {};
#pragma unroll
    for (int kk = 0; kk < 4; ++kk) {
      v16bf a = load_frag_nt(qsh, KLD, 0, kk * 32, lane);
      qk = __builtin_amdgcn_wmma_f32_16x16x32_bf16(false, a, false, bk[kk],
                                                   (short)0, qk, false, false);
    }
#pragma unroll
    for (int v = 0; v < 8; ++v) {
      float wgt = wsh[(rloc + v) * NH + h];
      float z = qk[v];
      accL[v] += (z > 0.0f ? z : 0.0f) * wgt;
    }
    __syncthreads();   // qsh rewritten next iteration
  }
  const int col = sC0 + w * 16 + (lane & 15);
#pragma unroll
  for (int v = 0; v < 8; ++v) {
    int row = t0 + rloc + v;
    float m = (col <= row) ? 0.0f : -1.0e9f;   // causal mask
    Lg[(size_t)row * TT + col] = accL[v] + m;
  }
}

// Per-row: bitonic sort 2048 (val,idx) descending in LDS, top-1024 + log-softmax.
__global__ void __launch_bounds__(256)
topk_kernel(const float* __restrict__ Lg, int* __restrict__ outIdx,
            float* __restrict__ outLp) {
  const int t = blockIdx.x;
  const int tid = threadIdx.x;
  __shared__ float sv[TT];
  __shared__ int   si[TT];
  __shared__ float red[256];
  for (int i = tid; i < TT; i += 256) { sv[i] = Lg[(size_t)t * TT + i]; si[i] = i; }
  __syncthreads();
  for (int ksz = 2; ksz <= TT; ksz <<= 1) {
    for (int j = ksz >> 1; j > 0; j >>= 1) {
      for (int i = tid; i < TT; i += 256) {
        int p = i ^ j;
        if (p > i) {
          bool desc = ((i & ksz) == 0);
          float a = sv[i], b = sv[p];
          bool sw = desc ? (a < b) : (a > b);
          if (sw) {
            sv[i] = b; sv[p] = a;
            int q = si[i]; si[i] = si[p]; si[p] = q;
          }
        }
      }
      __syncthreads();
    }
  }
  float m = sv[0];
  float part = 0.0f;
  for (int i = tid; i < TOPKK; i += 256) part += expf(sv[i] - m);
  red[tid] = part; __syncthreads();
  for (int s = 128; s > 0; s >>= 1) { if (tid < s) red[tid] += red[tid + s]; __syncthreads(); }
  float lse = logf(red[0]);
  for (int i = tid; i < TOPKK; i += 256) {
    outIdx[(size_t)t * TOPKK + i] = si[i];
    outLp[(size_t)t * TOPKK + i] = sv[i] - m - lse;
  }
}

static inline size_t alignup(size_t x) { return (x + 255) & ~((size_t)255); }

extern "C" void kernel_launch(void* const* d_in, const int* in_sizes, int n_in,
                              void* d_out, int out_size, void* d_ws, size_t ws_size,
                              hipStream_t stream) {
  (void)in_sizes; (void)n_in; (void)out_size; (void)ws_size;
  const float* x     = (const float*)d_in[0];
  const float* qr    = (const float*)d_in[1];
  const float* freqs = (const float*)d_in[2];
  // d_in[3] = attention_mask (recomputed inline)
  const float* wq_b  = (const float*)d_in[4];
  const float* wk    = (const float*)d_in[5];
  const float* gamma = (const float*)d_in[6];
  const float* beta  = (const float*)d_in[7];
  const float* wproj = (const float*)d_in[8];

  char* ws = (char*)d_ws;
  size_t off = 0;
  auto alloc = [&](size_t bytes) { char* p = ws + off; off = alignup(off + bytes); return p; };
  unsigned short* x_bf   = (unsigned short*)alloc((size_t)TT * HIDN * 2);
  unsigned short* qr_bf  = (unsigned short*)alloc((size_t)TT * RR * 2);
  unsigned short* wqb_bf = (unsigned short*)alloc((size_t)NH * DD * RR * 2);
  unsigned short* wk_bf  = (unsigned short*)alloc((size_t)DD * HIDN * 2);
  unsigned short* wpj_bf = (unsigned short*)alloc((size_t)NH * HIDN * 2);
  float*          qraw   = (float*)alloc((size_t)TT * NH * DD * 4);
  float*          kraw   = (float*)alloc((size_t)TT * DD * 4);
  unsigned short* q_bf   = (unsigned short*)alloc((size_t)TT * NH * DD * 2);
  unsigned short* k_bf   = (unsigned short*)alloc((size_t)TT * DD * 2);
  float*          wts    = (float*)alloc((size_t)TT * NH * 4);
  float*          logits = (float*)alloc((size_t)TT * TT * 4);

  auto cvt = [&](const float* in, unsigned short* out, int n) {
    cvt_f32_bf16<<<(n + 255) / 256, 256, 0, stream>>>(in, out, n);
  };
  cvt(x,     x_bf,   TT * HIDN);
  cvt(qr,    qr_bf,  TT * RR);
  cvt(wq_b,  wqb_bf, NH * DD * RR);
  cvt(wk,    wk_bf,  DD * HIDN);
  cvt(wproj, wpj_bf, NH * HIDN);

  // q-proj: [2048 x 4096] = qr[2048 x 1536] @ wq_b[4096 x 1536]^T
  gemm_nt_bf16_wmma<<<dim3((NH * DD) / 16, TT / 16), 32, 0, stream>>>(
      qr_bf, wqb_bf, qraw, NH * DD, RR, 1.0f);
  // k-proj: [2048 x 128] = x[2048 x 4096] @ wk[128 x 4096]^T
  gemm_nt_bf16_wmma<<<dim3(DD / 16, TT / 16), 32, 0, stream>>>(
      x_bf, wk_bf, kraw, DD, HIDN, 1.0f);
  // head weights: [2048 x 32] = x @ wproj^T * H^-0.5
  gemm_nt_bf16_wmma<<<dim3(NH / 16, TT / 16), 32, 0, stream>>>(
      x_bf, wpj_bf, wts, NH, HIDN, 0.1767766952966369f);

  q_post<<<TT, 256, 0, stream>>>(qraw, freqs, q_bf);
  k_post<<<TT, 128, 0, stream>>>(kraw, freqs, gamma, beta, k_bf);

  logits_wmma<<<dim3(TT / 128, TT / 16), 256, 0, stream>>>(q_bf, k_bf, wts, logits);

  int*   oIdx = (int*)d_out;
  float* oLp  = (float*)d_out + (size_t)TT * TOPKK;
  topk_kernel<<<TT, 256, 0, stream>>>(logits, oIdx, oLp);
}
